// MultiSupConLoss_17076789969009
// MI455X (gfx1250) — compile-verified, matrix operand order checked by reading, not run on hardware
//
#include <hip/hip_runtime.h>
#include <hip/hip_bf16.h>

typedef __attribute__((ext_vector_type(16))) _Float16 v16h;
typedef __attribute__((ext_vector_type(8)))  _Float16 v8h;
typedef __attribute__((ext_vector_type(8)))  float    v8f;

#define B_SAMP 4096
#define NVIEW  2
#define DIM    128
#define NTOT   (B_SAMP * NVIEW)     // 8192
#define NTILES (NTOT / 16)          // 512
#define TEMP   0.07f
#define EPSV   1e-8f
#define CTH    0.3f

// ---------------------------------------------------------------------------
// Prep: transpose features (B,V,D)->(V*B,D) converting fp32->f16,
// pack 6 binary labels per sample into a bitmask, zero the output scalar.
// ---------------------------------------------------------------------------
__global__ void prep_kernel(const float* __restrict__ features,
                            const int* __restrict__ labels,
                            _Float16* __restrict__ feat16,
                            unsigned int* __restrict__ lb,
                            float* __restrict__ out) {
    int idx = blockIdx.x * blockDim.x + threadIdx.x;
    if (idx < NTOT * DIM) {
        int n = idx >> 7;            // row in contrast order, n = v*B + b
        int d = idx & (DIM - 1);
        int v = n >> 12;             // / B_SAMP
        int b = n & (B_SAMP - 1);
        feat16[idx] = (_Float16)features[(b * NVIEW + v) * DIM + d];
    }
    if (idx < B_SAMP) {
        unsigned int m = 0;
#pragma unroll
        for (int k = 0; k < 6; ++k)
            m |= ((unsigned int)labels[idx * 6 + k] & 1u) << k;
        lb[idx] = m;
    }
    if (idx == 0) *out = 0.0f;
}

// Load one 16x16 B tile (4 K-step fragments). K-run layout per the 16-bit
// WMMA operand spec: lanes 0-15 hold K {0..7,16..23}, lanes 16-31 {8..15,24..31}
// -> two aligned 16B loads per fragment (compiler emits global_load_b128).
__device__ __forceinline__ void load_btile(const _Float16* __restrict__ base,
                                           v16h bfr[4]) {
#pragma unroll
    for (int kk = 0; kk < 4; ++kk) {
        v8h lo = *(const v8h*)(base + kk * 32);
        v8h hi = *(const v8h*)(base + kk * 32 + 16);
#pragma unroll
        for (int t = 0; t < 8; ++t) { bfr[kk][t] = lo[t]; bfr[kk][t + 8] = hi[t]; }
    }
}

// ---------------------------------------------------------------------------
// Fused single-pass streaming loss: one wave per 16-row anchor block.
// Online (flash-style) max/exp-sum; numerator uses raw logits so only the
// exp-sum needs rescaling. Double-buffered B tiles; label weights via LDS LUT.
// ---------------------------------------------------------------------------
__global__ void __launch_bounds__(32)
loss_kernel(const _Float16* __restrict__ feat,
            const unsigned int* __restrict__ lb,
            float* __restrict__ out) {
    __shared__ float2 tab[64 * 64];          // {w, w*multi} per label-mask pair

    const int lane   = threadIdx.x & 31;
    const int lane15 = lane & 15;
    const int lane16 = lane >> 4;
    const int i0     = blockIdx.x * 16;
    const float invT = 1.0f / (TEMP + EPSV);

    // Build 64x64 label LUT once per block (labels are 6-bit masks).
    for (int idx = lane; idx < 64 * 64; idx += 32) {
        unsigned int li = (unsigned)idx >> 6, lj = (unsigned)idx & 63u;
        float up = (float)__popc(li & lj);
        float lo = (float)__popc(li | lj);
        float multi = up / (lo + EPSV);
        float w = (multi >= CTH) ? 1.0f : 0.0f;
        tab[idx] = make_float2(w, w * multi);
    }
    __syncthreads();

    // Label-row indices (C/D layout: row M = r + 8*lane16), pre-shifted.
    int lbrow6[8];
#pragma unroll
    for (int r = 0; r < 8; ++r)
        lbrow6[r] = (int)(lb[(i0 + r + 8 * lane16) & (B_SAMP - 1)] << 6);

    // A fragments are loop-invariant: hoist all 4 K-steps into registers.
    const _Float16* arow = feat + (size_t)(i0 + lane15) * DIM + lane16 * 8;
    v16h afrag[4];
    load_btile(arow, afrag);                 // same fragment layout as B

    float vmax[8], acc_e[8], acc_a[8], acc_b[8], acc_c[8];
#pragma unroll
    for (int r = 0; r < 8; ++r) {
        vmax[r] = -3.0e38f;
        acc_e[r] = 0.f; acc_a[r] = 0.f; acc_b[r] = 0.f; acc_c[r] = 0.f;
    }

    const _Float16* bbase = feat + (size_t)lane15 * DIM + lane16 * 8;
    const v8f zero = {};

    auto process = [&](int J, const v16h bfr[4]) {
        // Two independent accumulation chains for WMMA ILP.
        v8f c0 = __builtin_amdgcn_wmma_f32_16x16x32_f16(
                     false, afrag[0], false, bfr[0], (short)0, zero, false, false);
        v8f c1 = __builtin_amdgcn_wmma_f32_16x16x32_f16(
                     false, afrag[1], false, bfr[1], (short)0, zero, false, false);
        c0 = __builtin_amdgcn_wmma_f32_16x16x32_f16(
                     false, afrag[2], false, bfr[2], (short)0, c0, false, false);
        c1 = __builtin_amdgcn_wmma_f32_16x16x32_f16(
                     false, afrag[3], false, bfr[3], (short)0, c1, false, false);

        const int j = J * 16 + lane15;
        const unsigned int lbj = lb[j & (B_SAMP - 1)];
#pragma unroll
        for (int r = 0; r < 8; ++r) {
            const float s = (c0[r] + c1[r]) * invT;
            const int   i = i0 + r + 8 * lane16;
            const float notdiag = (i == j) ? 0.0f : 1.0f;
            const float2 t = tab[lbrow6[r] | (int)lbj];
            // online lane-local softmax accumulator
            const float mn = fmaxf(vmax[r], s);
            acc_e[r] = acc_e[r] * __expf(vmax[r] - mn) + notdiag * __expf(s - mn);
            vmax[r] = mn;
            const float w  = t.x * notdiag;
            const float wm = t.y * notdiag;
            acc_a[r] += wm * s;     // raw logits; M and log-sum applied at the end
            acc_b[r] += wm;
            acc_c[r] += w;
        }
    };

    // Ping-pong double buffer: next tile's loads fly over current tile's math.
    v16h buf0[4], buf1[4];
    load_btile(bbase, buf0);
    for (int J = 0; J < NTILES; J += 2) {
        load_btile(bbase + (size_t)(J + 1) * 16 * DIM, buf1);
        process(J, buf0);
        if (J + 2 < NTILES)
            load_btile(bbase + (size_t)(J + 2) * 16 * DIM, buf0);
        process(J + 1, buf1);
    }

    // Cross-lane combine within each 16-lane half (row data replicated after).
#pragma unroll
    for (int r = 0; r < 8; ++r) {
        float m = vmax[r], e = acc_e[r];
#pragma unroll
        for (int mx = 1; mx < 16; mx <<= 1) {
            const float mo = __shfl_xor(m, mx, 16);
            const float eo = __shfl_xor(e, mx, 16);
            const float mn = fmaxf(m, mo);
            e = e * __expf(m - mn) + eo * __expf(mo - mn);
            m = mn;
            acc_a[r] += __shfl_xor(acc_a[r], mx, 16);
            acc_b[r] += __shfl_xor(acc_b[r], mx, 16);
            acc_c[r] += __shfl_xor(acc_c[r], mx, 16);
        }
        vmax[r] = m; acc_e[r] = e;
    }

    if (lane15 == 0) {   // lanes 0 (rows 0..7) and 16 (rows 8..15)
        float partial = 0.0f;
#pragma unroll
        for (int r = 0; r < 8; ++r) {
            const float L = __logf(acc_e[r] + EPSV);
            const float mean = (acc_a[r] - acc_b[r] * (vmax[r] + L)) /
                               (acc_c[r] + EPSV);
            partial -= mean;                 // T/BASE_T == 1
        }
        atomicAdd(out, partial * (1.0f / (float)NTOT));
    }
}

// ---------------------------------------------------------------------------
extern "C" void kernel_launch(void* const* d_in, const int* in_sizes, int n_in,
                              void* d_out, int out_size, void* d_ws, size_t ws_size,
                              hipStream_t stream) {
    const float* features = (const float*)d_in[0];
    const int*   labels   = (const int*)d_in[1];
    float*       out      = (float*)d_out;

    _Float16*     feat16 = (_Float16*)d_ws;                                   // 2 MB
    unsigned int* lbmask = (unsigned int*)((char*)d_ws +
                             (size_t)NTOT * DIM * sizeof(_Float16));          // 16 KB

    const int total = NTOT * DIM;
    prep_kernel<<<(total + 255) / 256, 256, 0, stream>>>(features, labels,
                                                         feat16, lbmask, out);
    loss_kernel<<<NTILES, 32, 0, stream>>>(feat16, lbmask, out);
}
